// DeformConv3d_58368605553085
// MI455X (gfx1250) — matrix operand order repaired and checked by reading, not compiled, not run on hardware
//
#include <hip/hip_runtime.h>
#include <hip/hip_bf16.h>

// DeformConv3d (B=2, C=32, 24^3, k=3^3, Cout=64, stride/pad/dil=1, groups=1)
//
// Pipeline:
//   1) cvt_weight    : f32 weight [64][864] -> f16 in ws
//   2) sample_kernel : trilinear im2col -> colsT[b][n][k] f16 in ws (K contiguous)
//   3) gemm_wmma     : out = W x colsT via v_wmma_f32_16x16x32_f16,
//                      LDS-staged A/B tiles, triple-buffered async global->LDS copies.

typedef __attribute__((ext_vector_type(16))) _Float16 v16h;
typedef __attribute__((ext_vector_type(8)))  _Float16 v8h;
typedef __attribute__((ext_vector_type(8)))  float    v8f;
typedef __attribute__((ext_vector_type(4)))  int      v4i;

#define B_    2
#define C_    32
#define DHW   24
#define NSP   (24 * 24 * 24)   // 13824
#define KN    27
#define KTOT  (C_ * KN)        // 864
#define COUT  64
#define STEPS (KTOT / 32)      // 27
#define COLS_OFF 131072

#define AS1 __attribute__((address_space(1)))
#define AS3 __attribute__((address_space(3)))

// ---- async global->LDS copy of 16 bytes per lane -------------------------
__device__ __forceinline__ void cp16(const _Float16* g, _Float16* l) {
#if __has_builtin(__builtin_amdgcn_global_load_async_to_lds_b128)
    __builtin_amdgcn_global_load_async_to_lds_b128((AS1 v4i*)g, (AS3 v4i*)l,
                                                   /*offset=*/0, /*cpol=*/0);
#else
    *(v8h*)l = *(const v8h*)g;   // sync fallback (ordered by __syncthreads)
#endif
}

__device__ __forceinline__ void wait_async_le3() {
#if __has_builtin(__builtin_amdgcn_global_load_async_to_lds_b128)
#if __has_builtin(__builtin_amdgcn_s_wait_asynccnt)
    __builtin_amdgcn_s_wait_asynccnt(3);
#else
    asm volatile("s_wait_asynccnt 0x3" ::: "memory");
#endif
#endif
}
__device__ __forceinline__ void wait_async_le0() {
#if __has_builtin(__builtin_amdgcn_global_load_async_to_lds_b128)
#if __has_builtin(__builtin_amdgcn_s_wait_asynccnt)
    __builtin_amdgcn_s_wait_asynccnt(0);
#else
    asm volatile("s_wait_asynccnt 0x0" ::: "memory");
#endif
#endif
}

// --------------------------------------------------------------------------
__global__ __launch_bounds__(256) void cvt_weight(const float* __restrict__ w,
                                                  _Float16* __restrict__ wh) {
    int i = blockIdx.x * 256 + threadIdx.x;
    if (i < COUT * KTOT) wh[i] = (_Float16)w[i];
}

// One thread per (b, tap, n): trilinear sample for all 32 channels.
__global__ __launch_bounds__(256) void sample_kernel(const float* __restrict__ x,
                                                     const float* __restrict__ off,
                                                     _Float16* __restrict__ colsT) {
    int tid = blockIdx.x * 256 + threadIdx.x;
    const int total = B_ * KN * NSP;
    if (tid >= total) return;
    int n = tid % NSP;
    int t = tid / NSP;
    int k = t % KN;
    int b = t / KN;

    int ox = n % DHW, oy = (n / DHW) % DHW, oz = n / (DHW * DHW);
    int kx = k % 3,   ky = (k / 3) % 3,    kz = k / 9;

    const float* ob = off + ((size_t)b * (3 * KN) + k * 3) * NSP + n;
    float z  = (float)(oz - 1 + kz) + ob[0];
    float y  = (float)(oy - 1 + ky) + ob[NSP];
    float xx = (float)(ox - 1 + kx) + ob[2 * NSP];

    float z0f = floorf(z), y0f = floorf(y), x0f = floorf(xx);
    float fz = z - z0f, fy = y - y0f, fx = xx - x0f;
    int z0 = (int)z0f, y0 = (int)y0f, x0 = (int)x0f;

    float acc[C_];
#pragma unroll
    for (int c = 0; c < C_; ++c) acc[c] = 0.0f;

    const float* xb = x + (size_t)b * C_ * NSP;
#pragma unroll
    for (int dz = 0; dz < 2; ++dz) {
        int iz = z0 + dz;
        float wz = dz ? fz : 1.0f - fz;
        if (iz < 0 || iz >= DHW) continue;
#pragma unroll
        for (int dy = 0; dy < 2; ++dy) {
            int iy = y0 + dy;
            float wy = dy ? fy : 1.0f - fy;
            if (iy < 0 || iy >= DHW) continue;
#pragma unroll
            for (int dx = 0; dx < 2; ++dx) {
                int ix = x0 + dx;
                if (ix < 0 || ix >= DHW) continue;
                float wc = (dx ? fx : 1.0f - fx) * wy * wz;
                const float* xp = xb + (iz * DHW + iy) * DHW + ix;
#pragma unroll
                for (int c = 0; c < C_; ++c)
                    acc[c] = fmaf(xp[(size_t)c * NSP], wc, acc[c]);
            }
        }
    }

    _Float16* dst = colsT + ((size_t)b * NSP + n) * KTOT + k;
#pragma unroll
    for (int c = 0; c < C_; ++c) dst[(size_t)c * KN] = (_Float16)acc[c];
}

// Block = 8 waves computing a 64(M) x 128(N) strip of one batch.
// wave: mt = wave&3 (M tile), nhalf = wave>>2 (which 64 N columns).
// Triple-buffered LDS tiles, async copies overlap with WMMA of previous step.
__global__ __launch_bounds__(256) void gemm_wmma(const _Float16* __restrict__ wh,
                                                 const _Float16* __restrict__ colsT,
                                                 const float* __restrict__ bias,
                                                 float* __restrict__ out) {
    __shared__ __align__(16) _Float16 lA[3][64 * 32];    // [m][k32]
    __shared__ __align__(16) _Float16 lB[3][128 * 32];   // [n128][k32]

    const int t    = threadIdx.x;
    const int lane = t & 31;
    const int wave = t >> 5;
    const int NG   = NSP / 128;                 // 108 column-groups per batch
    const int b    = blockIdx.x / NG;
    const int ng   = blockIdx.x % NG;

    const int mt    = wave & 3;
    const int nhalf = wave >> 2;
    const int l15   = lane & 15;
    const int hiSel = lane >> 4;
    const int ahalf = hiSel << 3;               // A: K halves at {h..h+7, h+16..h+23}
    const int bhalf = hiSel << 3;               // B: K halves at {h..h+7, h+16..h+23}

    const _Float16* colsBase = colsT + ((size_t)b * NSP + (size_t)ng * 128) * KTOT;

    // copy-chunk assignment (16B per lane):
    const int am  = t >> 2,  ac = (t & 3) * 8;          // A: 64 rows x 4 chunks
    const int br0 = t >> 2,  bc0 = (t & 3) * 8;         // B: chunks 0..255
    const int br1 = (t + 256) >> 2, bc1 = ((t + 256) & 3) * 8;  // B: chunks 256..511

    // ---- prologue: issue copies for step 0 into buffer 0 ----
    {
        const int kk = 0;
        cp16(wh + (size_t)am * KTOT + kk + ac, &lA[0][am * 32 + ac]);
        cp16(colsBase + (size_t)br0 * KTOT + kk + bc0, &lB[0][br0 * 32 + bc0]);
        cp16(colsBase + (size_t)br1 * KTOT + kk + bc1, &lB[0][br1 * 32 + bc1]);
    }

    v8f acc0 = {}, acc1 = {}, acc2 = {}, acc3 = {};

    const int amRow = (mt * 16 + l15) * 32;
    const int brRow0 = (nhalf * 64 + l15) * 32;

    for (int s = 0; s < STEPS; ++s) {
        const int buf = s % 3;
        if (s + 1 < STEPS) {            // issue next step into buffer (s+1)%3
            const int nb = (s + 1) % 3;
            const int kk = (s + 1) * 32;
            cp16(wh + (size_t)am * KTOT + kk + ac, &lA[nb][am * 32 + ac]);
            cp16(colsBase + (size_t)br0 * KTOT + kk + bc0, &lB[nb][br0 * 32 + bc0]);
            cp16(colsBase + (size_t)br1 * KTOT + kk + bc1, &lB[nb][br1 * 32 + bc1]);
            wait_async_le3();           // step-s copies complete, s+1 in flight
        } else {
            wait_async_le0();
        }
        __syncthreads();                // step-s tiles visible block-wide

        // A fragment from LDS
        v8h alo = *(const v8h*)&lA[buf][amRow + ahalf];
        v8h ahi = *(const v8h*)&lA[buf][amRow + ahalf + 16];
        v16h a = __builtin_shufflevector(alo, ahi,
                 0,1,2,3,4,5,6,7,8,9,10,11,12,13,14,15);

#pragma unroll
        for (int sub = 0; sub < 4; ++sub) {
            const int row = brRow0 + sub * 16 * 32;
            v8h blo = *(const v8h*)&lB[buf][row + bhalf];
            v8h bhi = *(const v8h*)&lB[buf][row + bhalf + 16];
            v16h bm = __builtin_shufflevector(blo, bhi,
                      0,1,2,3,4,5,6,7,8,9,10,11,12,13,14,15);
            v8f& acc = (sub == 0) ? acc0 : (sub == 1) ? acc1 : (sub == 2) ? acc2 : acc3;
            acc = __builtin_amdgcn_wmma_f32_16x16x32_f16(
                false, a, false, bm, (short)0, acc, false, false);
        }
    }

    // C/D layout: element rr <-> (m = rr + 8*hiSel, n = lane&15)
    const int mrow = mt * 16 + (hiSel << 3);
#pragma unroll
    for (int sub = 0; sub < 4; ++sub) {
        const v8f& acc = (sub == 0) ? acc0 : (sub == 1) ? acc1 : (sub == 2) ? acc2 : acc3;
        int ncol = ng * 128 + nhalf * 64 + sub * 16 + l15;
        float* op = out + ((size_t)b * COUT + mrow) * NSP + ncol;
#pragma unroll
        for (int rr = 0; rr < 8; ++rr)
            op[(size_t)rr * NSP] = acc[rr] + bias[mrow + rr];
    }
}

extern "C" void kernel_launch(void* const* d_in, const int* in_sizes, int n_in,
                              void* d_out, int out_size, void* d_ws, size_t ws_size,
                              hipStream_t stream) {
    const float* x    = (const float*)d_in[0];
    const float* off  = (const float*)d_in[1];
    const float* w    = (const float*)d_in[2];
    const float* bias = (const float*)d_in[3];
    float* out = (float*)d_out;

    _Float16* wh    = (_Float16*)d_ws;
    _Float16* colsT = (_Float16*)((char*)d_ws + COLS_OFF);

    cvt_weight<<<(COUT * KTOT + 255) / 256, 256, 0, stream>>>(w, wh);
    sample_kernel<<<(B_ * KN * NSP + 255) / 256, 256, 0, stream>>>(x, off, colsT);
    gemm_wmma<<<B_ * (NSP / 128), 256, 0, stream>>>(wh, colsT, bias, out);
}
